// Linear4bit_85014582657559
// MI455X (gfx1250) — matrix operand order repaired and checked by reading, not compiled, not run on hardware
//
#include <hip/hip_runtime.h>

// ---- problem constants (match reference) ----
#define BATCH 8
#define IN_F 4096
#define OUT_F 16384
#define WPR (IN_F / 2)     // packed int32 words per weight row (one byte payload each)
#define KSPLIT 4           // waves per tile (K-split)
#define KCHUNK (IN_F / KSPLIT)

typedef __attribute__((ext_vector_type(16))) _Float16 v16h;
typedef __attribute__((ext_vector_type(8)))  _Float16 v8h;
typedef __attribute__((ext_vector_type(8)))  float    v8f;
typedef __attribute__((ext_vector_type(2)))  _Float16 h2;

// Dequant one packed byte (low 8 bits of an int32 word) into a packed half2:
//   half[0] = signed int4 from low nibble, half[1] = signed int4 from high nibble.
// nibble^8 = signed+8; 0x6408 = f16(1032); placing (nibble^8) in the mantissa of
// 0x6400 gives 1032+s exactly; one v_pk_add_f16 of -1032 recovers s exactly.
__device__ __forceinline__ h2 dq_pair(int w) {
    int t = (w & 0xF) | ((w << 12) & 0xF0000);
    t ^= 0x64086408;                       // folds to v_bitop3_b32
    h2 m = __builtin_bit_cast(h2, t);
    const h2 off = {(_Float16)(-1032.0f), (_Float16)(-1032.0f)};
    return m + off;                        // v_pk_add_f16
}

union BFrag { v16h v; h2 p[8]; };
union AccU  { v8f v; float4 q[2]; };

__global__ void __launch_bounds__(128)
w4a16_wmma_kernel(const _Float16* __restrict__ x,        // [8, 4096] fp16
                  const int*      __restrict__ wq,       // [16384, 2048] int32 (1 byte payload)
                  const _Float16* __restrict__ scales,   // [16384] fp16
                  _Float16*       __restrict__ out)      // [8, 16384] fp16
{
    const int lane   = threadIdx.x & 31;
    const int wave   = threadIdx.x >> 5;   // 0..3 : K-split index
    const int tile   = blockIdx.x;         // 0..1023 : 16 output channels
    const int col    = lane & 15;
    const int hiHalf = lane >> 4;          // 0: lanes 0-15, 1: lanes 16-31

    // B fragment: lane owns channel o; K chunks {0|8} and +16 -> word base 0/4 and +8.
    // Lanes l and l+16 cover disjoint K halves, waves cover disjoint K quarters
    // => each weight word is loaded exactly once overall.
    const int    o      = tile * 16 + col;
    const int*   wrow   = wq + (size_t)o * WPR;
    const int    kwBase = hiHalf * 4;

    // A fragment: lane owns x row m; K chunks at base {0|8} and base+16.
    // Pad rows 8..15 DUPLICATE rows 0..7 (m&7); their D rows are discarded,
    // so no zero-fill and no divergence.
    const int    aBase  = hiHalf * 8;
    const _Float16* xrow = x + (size_t)(col & 7) * IN_F;

    const int k0 = wave * KCHUNK;
    v8f acc = {};

#pragma unroll 4
    for (int k = k0; k < k0 + KCHUNK; k += 32) {
        // ---- B: words kw..kw+3 and kw+8..kw+11 (16 K values), two b128 loads ----
        const int kw = (k >> 1) + kwBase;
        const int4 w0 = *(const int4*)(wrow + kw);
        const int4 w1 = *(const int4*)(wrow + kw + 8);

        // ---- A: two 16B chunks of x row (col&7), unconditional ----
        const v8h a0 = *(const v8h*)(xrow + k + aBase);        // K = k+aBase .. +7
        const v8h a1 = *(const v8h*)(xrow + k + aBase + 16);   // K = k+aBase+16 .. +23
        const v16h a = __builtin_shufflevector(a0, a1, 0, 1, 2, 3, 4, 5, 6, 7,
                                                       8, 9, 10, 11, 12, 13, 14, 15);

        BFrag bf;
        bf.p[0] = dq_pair(w0.x); bf.p[1] = dq_pair(w0.y);
        bf.p[2] = dq_pair(w0.z); bf.p[3] = dq_pair(w0.w);
        bf.p[4] = dq_pair(w1.x); bf.p[5] = dq_pair(w1.y);
        bf.p[6] = dq_pair(w1.z); bf.p[7] = dq_pair(w1.w);

        // D = A(16x32) * B(32x16) + C
        acc = __builtin_amdgcn_wmma_f32_16x16x32_f16(
                  /*neg_a=*/false, a, /*neg_b=*/false, bf.v,
                  /*c_mod=*/(short)0, acc, /*reuse_a=*/false, /*reuse_b=*/false);
    }

    // ---- K-split reduction through LDS (waves 1..3 -> wave 0) ----
    __shared__ float4 red[KSPLIT * 32 * 2];   // 4 KB
    if (wave != 0) {
        AccU u; u.v = acc;
        red[(wave * 32 + lane) * 2 + 0] = u.q[0];
        red[(wave * 32 + lane) * 2 + 1] = u.q[1];
    }
    __syncthreads();

    if (wave == 0) {
#pragma unroll
        for (int w = 1; w < KSPLIT; ++w) {
            const float4 r0 = red[(w * 32 + lane) * 2 + 0];
            const float4 r1 = red[(w * 32 + lane) * 2 + 1];
            acc[0] += r0.x; acc[1] += r0.y; acc[2] += r0.z; acc[3] += r0.w;
            acc[4] += r1.x; acc[5] += r1.y; acc[6] += r1.z; acc[7] += r1.w;
        }
        // Epilogue: deferred per-channel scale; store real batch rows only.
        // C/D layout: lanes 0-15 hold M=0..7 (VGPR r = row r) for column n = col.
        const float s = (float)scales[o];
        if (hiHalf == 0) {
#pragma unroll
            for (int r = 0; r < 8; ++r) {
                out[(size_t)r * OUT_F + o] = (_Float16)(acc[r] * s);
            }
        }
    }
}

extern "C" void kernel_launch(void* const* d_in, const int* in_sizes, int n_in,
                              void* d_out, int out_size, void* d_ws, size_t ws_size,
                              hipStream_t stream) {
    const _Float16* x      = (const _Float16*)d_in[0];
    const int*      wq     = (const int*)d_in[1];
    const _Float16* scales = (const _Float16*)d_in[2];
    _Float16*       out    = (_Float16*)d_out;

    // One 16-channel tile per 128-thread block (4 K-split waves) -> 1024 blocks,
    // 4096 waves: fills a large-WGP-count part far better than 128 fat blocks.
    dim3 grid(OUT_F / 16);
    dim3 block(32 * KSPLIT);
    w4a16_wmma_kernel<<<grid, block, 0, stream>>>(x, wq, scales, out);
}
